// TransformerDiffusion_50878182588507
// MI455X (gfx1250) — matrix operand order repaired and checked by reading, not compiled, not run on hardware
//
#include <hip/hip_runtime.h>
#include <math.h>

// Problem constants (from reference)
#define N_      1536
#define M_      768
#define D_      256
#define HEADS_  8
#define LAYERS_ 6
#define S_      (N_ + M_)      // 2304
#define DH_     (D_ / HEADS_)  // 32
#define FF_     (4 * D_)       // 1024

typedef __attribute__((ext_vector_type(16))) _Float16 v16h;
typedef __attribute__((ext_vector_type(8)))  _Float16 v8h;
typedef __attribute__((ext_vector_type(8)))  float    v8f;

// ---------------------------------------------------------------------------
// Prep: nodes = [|r_t| ; (H @ bits) mod 2],  bits = 0.5*(1-sign(r_t))
// ---------------------------------------------------------------------------
__global__ void nodes_kernel(const float* __restrict__ r_t,
                             const float* __restrict__ H,
                             float* __restrict__ nodes) {
    int i = blockIdx.x * blockDim.x + threadIdx.x;
    if (i >= S_) return;
    if (i < N_) {
        nodes[i] = fabsf(r_t[i]);
    } else {
        int j = i - N_;
        const float* hrow = H + (size_t)j * N_;
        float s = 0.f;
        for (int n = 0; n < N_; ++n) {
            float r   = r_t[n];
            float sgn = (r > 0.f) ? 1.f : ((r < 0.f) ? -1.f : 0.f);
            s += hrow[n] * (0.5f * (1.f - sgn));
        }
        nodes[i] = fmodf(s, 2.0f);
    }
}

// x[s,d] = src_embed[s,d] * nodes[s] * time_table[t,d]   (f32 + f16 mirror)
__global__ void embed_kernel(const float* __restrict__ src_embed,
                             const float* __restrict__ time_table,
                             const float* __restrict__ nodes,
                             const int*   __restrict__ t_ptr,
                             float* __restrict__ x,
                             _Float16* __restrict__ x16) {
    int idx = blockIdx.x * blockDim.x + threadIdx.x;
    if (idx >= S_ * D_) return;
    int s = idx / D_;
    int d = idx - s * D_;
    int t = t_ptr[0];
    float val = src_embed[idx] * nodes[s] * time_table[(size_t)t * D_ + d];
    x[idx]   = val;
    x16[idx] = (_Float16)val;
}

// Flat f32 -> f16 conversion (weights, done once per launch)
__global__ void cvt16_kernel(const float* __restrict__ src,
                             _Float16* __restrict__ dst, int n) {
    int i = blockIdx.x * blockDim.x + threadIdx.x;
    if (i < n) dst[i] = (_Float16)src[i];
}

// ---------------------------------------------------------------------------
// Fragment helpers (layouts per CDNA5 ISA 7.12.2, wave32):
//   A 16x32 f16 : lane L: row = L&15, K-base = (L>>4)*8; halves 0..7 -> K=kb+j,
//                 halves 8..15 -> K=kb+16+j
//   B 32x16 f16 : lane L: col = L&15, K-base = (L>>4)*8; same K split as A
//   C 16x16 f32 : lane L: col = L&15, rows = (L>>4)*8 + v (v = VGPR 0..7)
// ---------------------------------------------------------------------------
__device__ __forceinline__ v16h combine8(v8h lo, v8h hi) {
    return __builtin_shufflevector(lo, hi, 0, 1, 2, 3, 4, 5, 6, 7,
                                           8, 9, 10, 11, 12, 13, 14, 15);
}

// A fragment: two 8-half chunks at koff and koff+16 (each one b128 load)
__device__ __forceinline__ v16h load_a_frag(const _Float16* arow, int koff) {
    v8h lo = *(const v8h*)(arow + koff);
    v8h hi = *(const v8h*)(arow + koff + 16);
    return combine8(lo, hi);
}

// B fragment: one column; halves 0..7 -> K=kb+j, halves 8..15 -> K=kb+16+j (strided by ldb)
__device__ __forceinline__ v16h load_b_frag(const _Float16* brow, int ldb) {
    v16h r;
    #pragma unroll
    for (int j = 0; j < 8; ++j) {
        r[j]     = brow[(size_t)j * ldb];
        r[j + 8] = brow[(size_t)(16 + j) * ldb];
    }
    return r;
}

#define WMMA_F16(Acc, Af, Bf) \
    __builtin_amdgcn_wmma_f32_16x16x32_f16(false, (Af), false, (Bf), (short)0, (Acc), false, false)

// ---------------------------------------------------------------------------
// WMMA GEMM on f16 operands:  C[Mr x Nc] = epilogue( A[Mr x K] @ B[K x Nc] )
//   A element (m,k) at A[m*lda + k]      (f16)
//   B element (k,n) at B[k*ldb + n]      (f16, always contiguous in n)
//   epilogue: val = acc*outScale + bias[n] + maskAdd[m*maskLd+n]*maskScale,
//             optional ReLU; output to f32 C, f16 C, or f16 C transposed.
// One wave computes a 32x32 C macro-tile (2x2 register blocking = 4 WMMAs per
// K-step with A/B fragment reuse). K loop is software-pipelined: the loads for
// step k+32 are issued before the WMMAs of step k, so in-order VMEM completion
// lets the scheduler overlap 4 in-flight b128 A loads and 32 strided B loads with matrix work.
// 4 waves (128 threads) per block.
// Requires Mr%32==0, Nc%32==0, K%32==0 (true for all call sites).
// ---------------------------------------------------------------------------
__global__ __launch_bounds__(128) void gemm_wmma_kernel(
    const _Float16* __restrict__ A, int lda,
    const _Float16* __restrict__ B, int ldb,
    float* __restrict__ Cf32,
    _Float16* __restrict__ Cf16, int transC, int ldc,
    const float* __restrict__ bias,
    const float* __restrict__ maskAdd, int maskLd, float maskScale,
    float outScale, int doRelu,
    int Nc, int K)
{
    const int lane  = threadIdx.x & 31;
    const int wave  = threadIdx.x >> 5;
    const int tileN = blockIdx.x * 4 + wave;    // 32-column macro-tile index
    const int tileM = blockIdx.y;               // 32-row macro-tile index
    if (tileN * 32 >= Nc) return;               // uniform per wave
    const int colBase = tileN * 32;
    const int rowBase = tileM * 32;

    const int am = lane & 15;                   // A row within 16-row tile
    const int kb = (lane >> 4) * 8;             // A K-group base
    const _Float16* arow0 = A + (size_t)(rowBase + am) * lda;    // M tile 0
    const _Float16* arow1 = arow0 + (size_t)16 * lda;            // M tile 1
    const _Float16* bbase = B + colBase + am + (size_t)kb * ldb; // lane's column, K-group base

    v8f c00 = {}, c01 = {}, c10 = {}, c11 = {};

    // -------- pipeline prologue: fragments for k0 = 0 --------
    v16h a0 = load_a_frag(arow0, kb);
    v16h a1 = load_a_frag(arow1, kb);
    v16h b0 = load_b_frag(bbase, ldb);
    v16h b1 = load_b_frag(bbase + 16, ldb);

    for (int k0 = 32; k0 < K; k0 += 32) {
        // ---- issue next-step loads before consuming current fragments ----
        v16h na0 = load_a_frag(arow0, k0 + kb);
        v16h na1 = load_a_frag(arow1, k0 + kb);
        const _Float16* nbrow = bbase + (size_t)k0 * ldb;
        v16h nb0 = load_b_frag(nbrow, ldb);
        v16h nb1 = load_b_frag(nbrow + 16, ldb);

        // Prefetch two steps ahead (lowers to global_prefetch_b8)
        if (k0 + 32 < K) {
            __builtin_prefetch(arow0 + k0 + 32 + kb, 0, 3);
            __builtin_prefetch(arow1 + k0 + 32 + kb, 0, 3);
            __builtin_prefetch(nbrow + (size_t)32 * ldb, 0, 3);
        }

        c00 = WMMA_F16(c00, a0, b0);
        c01 = WMMA_F16(c01, a0, b1);
        c10 = WMMA_F16(c10, a1, b0);
        c11 = WMMA_F16(c11, a1, b1);

        a0 = na0; a1 = na1; b0 = nb0; b1 = nb1;
    }

    // -------- pipeline epilogue: last K-step --------
    c00 = WMMA_F16(c00, a0, b0);
    c01 = WMMA_F16(c01, a0, b1);
    c10 = WMMA_F16(c10, a1, b0);
    c11 = WMMA_F16(c11, a1, b1);

    // ---- epilogue + store (C layout per ISA) ----
    const int cn = lane & 15;
    const int mb = (lane >> 4) * 8;
    #pragma unroll
    for (int v = 0; v < 8; ++v) {
        int r0 = rowBase + mb + v;
        int r1 = r0 + 16;
        int n0 = colBase + cn;
        int n1 = n0 + 16;
        float v00 = c00[v] * outScale;
        float v01 = c01[v] * outScale;
        float v10 = c10[v] * outScale;
        float v11 = c11[v] * outScale;
        if (bias) {
            v00 += bias[n0]; v01 += bias[n1];
            v10 += bias[n0]; v11 += bias[n1];
        }
        if (maskAdd) {
            v00 += maskAdd[(size_t)r0 * maskLd + n0] * maskScale;
            v01 += maskAdd[(size_t)r0 * maskLd + n1] * maskScale;
            v10 += maskAdd[(size_t)r1 * maskLd + n0] * maskScale;
            v11 += maskAdd[(size_t)r1 * maskLd + n1] * maskScale;
        }
        if (doRelu) {
            v00 = fmaxf(v00, 0.f); v01 = fmaxf(v01, 0.f);
            v10 = fmaxf(v10, 0.f); v11 = fmaxf(v11, 0.f);
        }
        if (Cf32) {
            Cf32[(size_t)r0 * ldc + n0] = v00;
            Cf32[(size_t)r0 * ldc + n1] = v01;
            Cf32[(size_t)r1 * ldc + n0] = v10;
            Cf32[(size_t)r1 * ldc + n1] = v11;
        } else if (!transC) {
            Cf16[(size_t)r0 * ldc + n0] = (_Float16)v00;
            Cf16[(size_t)r0 * ldc + n1] = (_Float16)v01;
            Cf16[(size_t)r1 * ldc + n0] = (_Float16)v10;
            Cf16[(size_t)r1 * ldc + n1] = (_Float16)v11;
        } else {            // transposed f16 store (used for k^T)
            Cf16[(size_t)n0 * ldc + r0] = (_Float16)v00;
            Cf16[(size_t)n1 * ldc + r0] = (_Float16)v01;
            Cf16[(size_t)n0 * ldc + r1] = (_Float16)v10;
            Cf16[(size_t)n1 * ldc + r1] = (_Float16)v11;
        }
    }
}

// ---------------------------------------------------------------------------
// Row softmax over S_ columns (one block per row); f32 in, f16 probs out
// ---------------------------------------------------------------------------
__global__ void softmax_kernel(float* __restrict__ scores,
                               _Float16* __restrict__ probs) {
    __shared__ float red[256];
    const int tid = threadIdx.x;
    float* srow = scores + (size_t)blockIdx.x * S_;
    _Float16* prow = probs + (size_t)blockIdx.x * S_;

    float mx = -3.0e38f;
    for (int c = tid; c < S_; c += 256) mx = fmaxf(mx, srow[c]);
    red[tid] = mx; __syncthreads();
    for (int s = 128; s > 0; s >>= 1) {
        if (tid < s) red[tid] = fmaxf(red[tid], red[tid + s]);
        __syncthreads();
    }
    mx = red[0]; __syncthreads();

    float sum = 0.f;
    for (int c = tid; c < S_; c += 256) {
        float e = expf(srow[c] - mx);
        srow[c] = e;
        sum += e;
    }
    red[tid] = sum; __syncthreads();
    for (int s = 128; s > 0; s >>= 1) {
        if (tid < s) red[tid] += red[tid + s];
        __syncthreads();
    }
    float inv = 1.f / red[0];
    for (int c = tid; c < S_; c += 256)
        prow[c] = (_Float16)(srow[c] * inv);
}

// ---------------------------------------------------------------------------
// x = LayerNorm(x + y) * g + b  (f32 result + f16 mirror); one block per row
// ---------------------------------------------------------------------------
__global__ void add_ln_kernel(float* __restrict__ x,
                              const float* __restrict__ y,
                              const float* __restrict__ g,
                              const float* __restrict__ b,
                              _Float16* __restrict__ x16) {
    __shared__ float red[256];
    const int row = blockIdx.x;
    const int d   = threadIdx.x;
    const size_t idx = (size_t)row * D_ + d;

    float val = x[idx] + y[idx];
    red[d] = val; __syncthreads();
    for (int s = 128; s > 0; s >>= 1) {
        if (d < s) red[d] += red[d + s];
        __syncthreads();
    }
    float mu = red[0] * (1.0f / D_); __syncthreads();

    float diff = val - mu;
    red[d] = diff * diff; __syncthreads();
    for (int s = 128; s > 0; s >>= 1) {
        if (d < s) red[d] += red[d + s];
        __syncthreads();
    }
    float var = red[0] * (1.0f / D_);
    float outv = diff * rsqrtf(var + 1e-6f) * g[d] + b[d];
    x[idx]   = outv;
    x16[idx] = (_Float16)outv;
}

// ---------------------------------------------------------------------------
// out[i] = (x[i,:] . fc_w + fc_b) * to_n_w + to_n_b   for i in [0, N_)
// ---------------------------------------------------------------------------
__global__ void head_kernel(const float* __restrict__ x,
                            const float* __restrict__ fc_w,
                            const float* __restrict__ fc_b,
                            const float* __restrict__ to_n_w,
                            const float* __restrict__ to_n_b,
                            float* __restrict__ out) {
    __shared__ float red[256];
    const int row = blockIdx.x;   // 0..N_-1
    const int d   = threadIdx.x;
    red[d] = x[(size_t)row * D_ + d] * fc_w[d];
    __syncthreads();
    for (int s = 128; s > 0; s >>= 1) {
        if (d < s) red[d] += red[d + s];
        __syncthreads();
    }
    if (d == 0)
        out[row] = (red[0] + fc_b[0]) * to_n_w[0] + to_n_b[0];
}

// ---------------------------------------------------------------------------
// Host orchestration
// ---------------------------------------------------------------------------
extern "C" void kernel_launch(void* const* d_in, const int* in_sizes, int n_in,
                              void* d_out, int out_size, void* d_ws, size_t ws_size,
                              hipStream_t stream) {
    (void)in_sizes; (void)n_in; (void)out_size; (void)ws_size;

    const float* r_t        = (const float*)d_in[0];
    const float* H          = (const float*)d_in[1];
    const float* mask       = (const float*)d_in[2];
    const float* src_embed  = (const float*)d_in[3];
    const float* time_table = (const float*)d_in[4];
    const float* Wq         = (const float*)d_in[5];
    const float* Wk         = (const float*)d_in[6];
    const float* Wv         = (const float*)d_in[7];
    const float* Wo         = (const float*)d_in[8];
    const float* W1         = (const float*)d_in[9];
    const float* b1         = (const float*)d_in[10];
    const float* W2         = (const float*)d_in[11];
    const float* b2         = (const float*)d_in[12];
    const float* ln1_g      = (const float*)d_in[13];
    const float* ln1_b      = (const float*)d_in[14];
    const float* ln2_g      = (const float*)d_in[15];
    const float* ln2_b      = (const float*)d_in[16];
    const float* fc_w       = (const float*)d_in[17];
    const float* fc_b       = (const float*)d_in[18];
    const float* to_n_w     = (const float*)d_in[19];
    const float* to_n_b     = (const float*)d_in[20];
    const int*   t_ptr      = (const int*)d_in[21];
    float*       out        = (float*)d_out;

    // ---------------- workspace layout (128B-aligned slices) ----------------
    float* wsf = (float*)d_ws;
    size_t off = 0;   // in floats
    auto allocF = [&](size_t n) {
        float* p = wsf + off; off += (n + 31) & ~(size_t)31; return p;
    };
    auto allocH = [&](size_t n) {
        _Float16* p = (_Float16*)(wsf + off);
        off += ((n + 1) / 2 + 31) & ~(size_t)31; return p;
    };

    float*    nodes   = allocF(S_);
    float*    x       = allocF((size_t)S_ * D_);
    float*    tmp     = allocF((size_t)S_ * D_);
    float*    scores  = allocF((size_t)S_ * S_);
    _Float16* x16     = allocH((size_t)S_ * D_);
    _Float16* q16     = allocH((size_t)S_ * D_);
    _Float16* kT16    = allocH((size_t)D_ * S_);   // transposed: [D][S]
    _Float16* v16     = allocH((size_t)S_ * D_);
    _Float16* o16     = allocH((size_t)S_ * D_);
    _Float16* probs16 = allocH((size_t)S_ * S_);
    _Float16* ff16    = allocH((size_t)S_ * FF_);
    _Float16* wq16    = allocH((size_t)LAYERS_ * D_ * D_);
    _Float16* wk16    = allocH((size_t)LAYERS_ * D_ * D_);
    _Float16* wv16    = allocH((size_t)LAYERS_ * D_ * D_);
    _Float16* wo16    = allocH((size_t)LAYERS_ * D_ * D_);
    _Float16* w116    = allocH((size_t)LAYERS_ * D_ * FF_);
    _Float16* w216    = allocH((size_t)LAYERS_ * FF_ * D_);

    // ---------------- weight conversion (f32 -> f16, once per launch) -------
    auto cvt = [&](const float* src, _Float16* dst, int n) {
        cvt16_kernel<<<(n + 255) / 256, 256, 0, stream>>>(src, dst, n);
    };
    cvt(Wq, wq16, LAYERS_ * D_ * D_);
    cvt(Wk, wk16, LAYERS_ * D_ * D_);
    cvt(Wv, wv16, LAYERS_ * D_ * D_);
    cvt(Wo, wo16, LAYERS_ * D_ * D_);
    cvt(W1, w116, LAYERS_ * D_ * FF_);
    cvt(W2, w216, LAYERS_ * FF_ * D_);

    // ---------------- prep ----------------
    nodes_kernel<<<(S_ + 255) / 256, 256, 0, stream>>>(r_t, H, nodes);
    embed_kernel<<<(S_ * D_ + 255) / 256, 256, 0, stream>>>(
        src_embed, time_table, nodes, t_ptr, x, x16);

    const float scale = 1.0f / sqrtf((float)DH_);

    auto gemm = [&](const _Float16* A, int lda, const _Float16* B, int ldb,
                    float* Cf32, _Float16* Cf16, int transC, int ldc,
                    const float* bias, const float* mAdd, int mLd, float mScale,
                    float oScale, int relu, int Mr, int Nc, int K) {
        dim3 grid((Nc / 32 + 3) / 4, Mr / 32);
        gemm_wmma_kernel<<<grid, 128, 0, stream>>>(
            A, lda, B, ldb, Cf32, Cf16, transC, ldc, bias,
            mAdd, mLd, mScale, oScale, relu, Nc, K);
    };

    for (int i = 0; i < LAYERS_; ++i) {
        const _Float16* wq = wq16 + (size_t)i * D_ * D_;
        const _Float16* wk = wk16 + (size_t)i * D_ * D_;
        const _Float16* wv = wv16 + (size_t)i * D_ * D_;
        const _Float16* wo = wo16 + (size_t)i * D_ * D_;
        const _Float16* w1 = w116 + (size_t)i * D_ * FF_;
        const _Float16* w2 = w216 + (size_t)i * FF_ * D_;

        // q/k/v projections: [S,D] @ [D,D]  (k stored transposed -> kT16)
        gemm(x16, D_, wq, D_, nullptr, q16, 0, D_,
             nullptr, nullptr, 0, 0.f, 1.f, 0, S_, D_, D_);
        gemm(x16, D_, wk, D_, nullptr, kT16, 1, S_,
             nullptr, nullptr, 0, 0.f, 1.f, 0, S_, D_, D_);
        gemm(x16, D_, wv, D_, nullptr, v16, 0, D_,
             nullptr, nullptr, 0, 0.f, 1.f, 0, S_, D_, D_);

        for (int h = 0; h < HEADS_; ++h) {
            // scores = (q_h @ k_h^T) * scale + mask * (-1e9)
            gemm(q16 + h * DH_, D_,                    // A: [S, 32]
                 kT16 + (size_t)h * DH_ * S_, S_,      // B: kT rows h*32..h*32+31
                 scores, nullptr, 0, S_,
                 nullptr, mask, S_, -1e9f,
                 scale, 0, S_, S_, DH_);
            softmax_kernel<<<S_, 256, 0, stream>>>(scores, probs16);
            // o_h = attn @ v_h : [S,S] @ [S,32]
            gemm(probs16, S_, v16 + h * DH_, D_,
                 nullptr, o16 + h * DH_, 0, D_,
                 nullptr, nullptr, 0, 0.f, 1.f, 0, S_, DH_, S_);
        }

        // output projection + residual + LN1
        gemm(o16, D_, wo, D_, tmp, nullptr, 0, D_,
             nullptr, nullptr, 0, 0.f, 1.f, 0, S_, D_, D_);
        add_ln_kernel<<<S_, 256, 0, stream>>>(
            x, tmp, ln1_g + i * D_, ln1_b + i * D_, x16);

        // FFN: relu(x@W1+b1)@W2+b2, residual + LN2
        gemm(x16, D_, w1, FF_, nullptr, ff16, 0, FF_,
             b1 + (size_t)i * FF_, nullptr, 0, 0.f, 1.f, 1, S_, FF_, D_);
        gemm(ff16, FF_, w2, D_, tmp, nullptr, 0, D_,
             b2 + (size_t)i * D_, nullptr, 0, 0.f, 1.f, 0, S_, D_, FF_);
        add_ln_kernel<<<S_, 256, 0, stream>>>(
            x, tmp, ln2_g + i * D_, ln2_b + i * D_, x16);
    }

    head_kernel<<<N_, 256, 0, stream>>>(x, fc_w, fc_b, to_n_w, to_n_b, out);
}
